// HierarchicalTemporalEncoder_48438641164540
// MI455X (gfx1250) — compile-verified, matrix operand order checked by reading, not run on hardware
//
#include <hip/hip_runtime.h>
#include <hip/hip_bf16.h>
#include <math.h>

// ---------------------------------------------------------------------------
// HierarchicalTemporalEncoder forward for gfx1250 (MI455X, wave32, WMMA).
// Dense projections: LDS-staged, register-blocked v_wmma_f32_16x16x32_f16
// (f32 accumulate). GATv2 processed one head at a time (~134 MB workspace).
// ---------------------------------------------------------------------------

#define D_DIM   128
#define H_HEADS 8
#define C_DIM   128
#define NB_     500
#define TT_     96
#define NZ_     50
#define NR_     5
#define FD_     32
#define N_ASSET (NB_ * TT_)   // 48000
#define N_ZONE  (NZ_ * TT_)   // 4800
#define N_REG   (NR_ * TT_)   // 480
#define TD      (TT_ * D_DIM) // 12288

typedef __attribute__((ext_vector_type(16))) _Float16 v16h;
typedef __attribute__((ext_vector_type(8)))  _Float16 v8h;
typedef __attribute__((ext_vector_type(8)))  float    v8f;

// ----------------------------- utility ------------------------------------

__global__ void fill_f32_kernel(float* __restrict__ p, float v, int n) {
  int i = blockIdx.x * blockDim.x + threadIdx.x;
  if (i < n) p[i] = v;
}
__global__ void fill_u32_kernel(unsigned* __restrict__ p, unsigned v, int n) {
  int i = blockIdx.x * blockDim.x + threadIdx.x;
  if (i < n) p[i] = v;
}

__device__ __forceinline__ unsigned enc_f32(float x) {
  unsigned u = __float_as_uint(x);
  return (u & 0x80000000u) ? ~u : (u | 0x80000000u);
}
__device__ __forceinline__ float dec_f32(unsigned u) {
  unsigned v = (u & 0x80000000u) ? (u & 0x7FFFFFFFu) : ~u;
  return __uint_as_float(v);
}
__device__ __forceinline__ float gelu_exact(float v) {
  return 0.5f * v * (1.f + erff(v * 0.70710678118f));
}

// ----------------------------- WMMA GEMM -----------------------------------
// Y[M,N] = X[M,K] * W[N,K]^T + bias[N], optional exact GELU.
// Block: 256 threads (8 waves), tile 128(M) x 64(N).
// B strip (64x32 f16) staged in LDS once per k-step (shared by all 8 waves);
// each wave holds a 16x64 strip = 4 WMMA accumulators; A fragment loaded once
// per k-step and reused across the 4 WMMAs.
// Fragment layout (CDNA5 16-bit A/B): row/col = lane&15, kb = (lane>>4)*8,
// half i<8 -> K=kb+i ; half i>=8 -> K=kb+8+i.

#define GEMM_BM 128
#define GEMM_BN 64
#define BS_STRIDE 40   // halfs; padded so 16-lane b128 reads hit distinct banks

__global__ __launch_bounds__(256)
void gemm_wmma_kernel(const float* __restrict__ X, const float* __restrict__ W,
                      const float* __restrict__ bias, float* __restrict__ Y,
                      int M, int N, int K, int gelu_flag) {
  __shared__ _Float16 Bs[GEMM_BN * BS_STRIDE];
  const int tid  = threadIdx.x;
  const int lane = tid & 31;
  const int wave = tid >> 5;                         // 0..7
  const int n0 = blockIdx.x * GEMM_BN;
  const int m0 = blockIdx.y * GEMM_BM + wave * 16;
  const bool valid = (m0 < M);

  v8f acc[4];
#pragma unroll
  for (int t = 0; t < 4; ++t) acc[t] = (v8f){0.f,0.f,0.f,0.f,0.f,0.f,0.f,0.f};

  const int arow = m0 + (lane & 15);
  const int akb  = (lane >> 4) << 3;
  const float* aptr = X + (size_t)(valid ? arow : 0) * K + akb;

  // cooperative B tile mapping: thread -> (col, 8 consecutive k)
  const int bcol = tid >> 2;                          // 0..63
  const int bk   = (tid & 3) << 3;                    // 0,8,16,24
  const float* wrow = W + (size_t)(n0 + bcol) * K + bk;
  _Float16* bs_dst = &Bs[bcol * BS_STRIDE + bk];

  for (int k = 0; k < K; k += 32) {
    __syncthreads();
    {  // stage B strip (fp32 -> f16) into LDS
      const float* p = wrow + k;
      v8h hv;
#pragma unroll
      for (int i = 0; i < 8; ++i) hv[i] = (_Float16)p[i];
      *(v8h*)bs_dst = hv;
      if (k + 32 < K) __builtin_prefetch(p + 32, 0, 1);
    }
    __syncthreads();

    v16h a;
    if (valid) {
      const float* p = aptr + k;
#pragma unroll
      for (int i = 0; i < 8; ++i) a[i] = (_Float16)p[i];
#pragma unroll
      for (int i = 0; i < 8; ++i) a[8 + i] = (_Float16)p[16 + i];
      if (k + 32 < K) __builtin_prefetch(p + 32, 0, 1);
    } else {
#pragma unroll
      for (int i = 0; i < 16; ++i) a[i] = (_Float16)0.f;
    }

#pragma unroll
    for (int t = 0; t < 4; ++t) {
      const _Float16* bp = &Bs[(t * 16 + (lane & 15)) * BS_STRIDE + akb];
      v8h lo = *(const v8h*)bp;          // K = kb .. kb+7
      v8h hi = *(const v8h*)(bp + 16);   // K = kb+16 .. kb+23
      v16h b;
#pragma unroll
      for (int i = 0; i < 8; ++i) { b[i] = lo[i]; b[8 + i] = hi[i]; }
      acc[t] = __builtin_amdgcn_wmma_f32_16x16x32_f16(false, a, false, b,
                                                      (short)0, acc[t], false, false);
    }
  }

  if (!valid) return;
  const int mb = m0 + ((lane >> 4) << 3);
#pragma unroll
  for (int t = 0; t < 4; ++t) {
    int nn = n0 + t * 16 + (lane & 15);
    float bv = bias ? bias[nn] : 0.f;
#pragma unroll
    for (int r = 0; r < 8; ++r) {
      float v = acc[t][r] + bv;
      if (gelu_flag) v = gelu_exact(v);
      Y[(size_t)(mb + r) * N + nn] = v;
    }
  }
}

// ----------------------------- GATv2 edge kernels ---------------------------
// Edge list layout: src = ei[e], dst = ei[Eorig+e] for e<Eorig; self loop else.

__device__ __forceinline__ void edge_src_dst(const int* __restrict__ ei, int Eorig,
                                             int e, int& src, int& dst) {
  if (e < Eorig) { src = ei[e]; dst = ei[Eorig + e]; }
  else           { src = dst = e - Eorig; }
}

// wave per edge, lane owns channels 4l..4l+3 (b128 loads):
// logit[e] = sum_c leakyrelu(xl[src,c]+xr[dst,c]) * att[c]
__global__ void edge_logit_kernel(const float* __restrict__ xl, const float* __restrict__ xr,
                                  const float* __restrict__ att, const int* __restrict__ ei,
                                  int Eorig, int Etot, float* __restrict__ logit,
                                  unsigned* __restrict__ nmax) {
  int e = blockIdx.x * 4 + (threadIdx.x >> 5);
  if (e >= Etot) return;
  int lane = threadIdx.x & 31;
  int src, dst; edge_src_dst(ei, Eorig, e, src, dst);
  const float4 l4 = *(const float4*)(xl + (size_t)src * D_DIM + lane * 4);
  const float4 r4 = *(const float4*)(xr + (size_t)dst * D_DIM + lane * 4);
  const float4 a4 = *(const float4*)(att + lane * 4);
  float v0 = l4.x + r4.x, v1 = l4.y + r4.y, v2 = l4.z + r4.z, v3 = l4.w + r4.w;
  v0 = (v0 > 0.f) ? v0 : 0.2f * v0;
  v1 = (v1 > 0.f) ? v1 : 0.2f * v1;
  v2 = (v2 > 0.f) ? v2 : 0.2f * v2;
  v3 = (v3 > 0.f) ? v3 : 0.2f * v3;
  float s = v0 * a4.x + v1 * a4.y + v2 * a4.z + v3 * a4.w;
#pragma unroll
  for (int off = 16; off > 0; off >>= 1) s += __shfl_xor(s, off, 32);
  if (lane == 0) {
    logit[e] = s;
    atomicMax(&nmax[dst], enc_f32(s));
  }
}

// thread per edge: ex[e] = exp(logit - max[dst]); den[dst] += ex
__global__ void edge_exp_kernel(const float* __restrict__ logit, const unsigned* __restrict__ nmax,
                                const int* __restrict__ ei, int Eorig, int Etot,
                                float* __restrict__ ex, float* __restrict__ nden) {
  int e = blockIdx.x * blockDim.x + threadIdx.x;
  if (e >= Etot) return;
  int src, dst; edge_src_dst(ei, Eorig, e, src, dst);
  (void)src;
  float v = __expf(logit[e] - dec_f32(nmax[dst]));
  ex[e] = v;
  atomicAdd(&nden[dst], v);
}

// wave per edge: out[dst,c] += xl[src,c] * alpha / H
__global__ void edge_accum_kernel(const float* __restrict__ xl, const float* __restrict__ ex,
                                  const float* __restrict__ nden, const int* __restrict__ ei,
                                  int Eorig, int Etot, float* __restrict__ out) {
  int e = blockIdx.x * 4 + (threadIdx.x >> 5);
  if (e >= Etot) return;
  int lane = threadIdx.x & 31;
  int src, dst; edge_src_dst(ei, Eorig, e, src, dst);
  float alpha = ex[e] / (nden[dst] + 1e-16f) * (1.f / (float)H_HEADS);
  const float4 l4 = *(const float4*)(xl + (size_t)src * D_DIM + lane * 4);
  float* po = out + (size_t)dst * D_DIM + lane * 4;
  atomicAdd(po + 0, l4.x * alpha);
  atomicAdd(po + 1, l4.y * alpha);
  atomicAdd(po + 2, l4.z * alpha);
  atomicAdd(po + 3, l4.w * alpha);
}

// ----------------------------- LayerNorm (wave per row, D=128) --------------
// y = LN(in + in_bias? + pre?) * g + b ; optional GELU ; optional +post
__global__ void ln_kernel(const float* __restrict__ in, const float* __restrict__ in_bias,
                          const float* __restrict__ pre, const float* __restrict__ g,
                          const float* __restrict__ b, const float* __restrict__ post,
                          float* __restrict__ out, int rows, int gelu_flag) {
  int row = blockIdx.x * 4 + (threadIdx.x >> 5);
  if (row >= rows) return;
  int lane = threadIdx.x & 31;
  size_t base = (size_t)row * D_DIM + lane * 4;
  float4 v4 = *(const float4*)(in + base);
  if (in_bias) {
    float4 ib = *(const float4*)(in_bias + lane * 4);
    v4.x += ib.x; v4.y += ib.y; v4.z += ib.z; v4.w += ib.w;
  }
  if (pre) {
    float4 p4 = *(const float4*)(pre + base);
    v4.x += p4.x; v4.y += p4.y; v4.z += p4.z; v4.w += p4.w;
  }
  float s = v4.x + v4.y + v4.z + v4.w;
#pragma unroll
  for (int off = 16; off > 0; off >>= 1) s += __shfl_xor(s, off, 32);
  float mean = s * (1.f / (float)D_DIM);
  float d0 = v4.x - mean, d1 = v4.y - mean, d2 = v4.z - mean, d3 = v4.w - mean;
  float vs = d0 * d0 + d1 * d1 + d2 * d2 + d3 * d3;
#pragma unroll
  for (int off = 16; off > 0; off >>= 1) vs += __shfl_xor(vs, off, 32);
  float rstd = rsqrtf(vs * (1.f / (float)D_DIM) + 1e-5f);
  float4 g4 = *(const float4*)(g + lane * 4);
  float4 b4 = *(const float4*)(b + lane * 4);
  float y0 = d0 * rstd * g4.x + b4.x;
  float y1 = d1 * rstd * g4.y + b4.y;
  float y2 = d2 * rstd * g4.z + b4.z;
  float y3 = d3 * rstd * g4.w + b4.w;
  if (gelu_flag) { y0 = gelu_exact(y0); y1 = gelu_exact(y1); y2 = gelu_exact(y2); y3 = gelu_exact(y3); }
  if (post) {
    float4 p4 = *(const float4*)(post + base);
    y0 += p4.x; y1 += p4.y; y2 += p4.z; y3 += p4.w;
  }
  float4 o4 = { y0, y1, y2, y3 };
  *(float4*)(out + base) = o4;
}

// ----------------------------- pooling / gather ------------------------------

__global__ void count_kernel(const int* __restrict__ assign, int nb, float* __restrict__ cnt) {
  int i = blockIdx.x * blockDim.x + threadIdx.x;
  if (i < nb) atomicAdd(&cnt[assign[i]], 1.f);
}
__global__ void pool_accum_kernel(const float* __restrict__ h, const int* __restrict__ assign,
                                  float* __restrict__ s, int total) {
  int idx = blockIdx.x * blockDim.x + threadIdx.x;
  if (idx >= total) return;
  int outer = idx / TD;
  int rem = idx - outer * TD;
  atomicAdd(&s[(size_t)assign[outer] * TD + rem], h[idx]);
}
__global__ void pool_div_kernel(float* __restrict__ s, const float* __restrict__ cnt, int total) {
  int idx = blockIdx.x * blockDim.x + threadIdx.x;
  if (idx >= total) return;
  int z = idx / TD;
  s[idx] /= fmaxf(cnt[z], 1.f);
}
// out[idx] = up[assign[outer]*TD + rem] + skip[idx]  (assign==null -> group 0)
__global__ void gather_add_kernel(const float* __restrict__ up, const float* __restrict__ skip,
                                  const int* __restrict__ assign, float* __restrict__ out,
                                  int total) {
  int idx = blockIdx.x * blockDim.x + threadIdx.x;
  if (idx >= total) return;
  int outer = idx / TD;
  int rem = idx - outer * TD;
  int grp = assign ? assign[outer] : 0;
  out[idx] = up[(size_t)grp * TD + rem] + skip[idx];
}

// nation = mean over NR regions + sinusoidal posenc
__global__ void nation_init_kernel(const float* __restrict__ rskip, float* __restrict__ x) {
  int idx = blockIdx.x * blockDim.x + threadIdx.x;
  if (idx >= TD) return;
  float s = 0.f;
#pragma unroll
  for (int r = 0; r < NR_; ++r) s += rskip[r * TD + idx];
  s *= (1.f / (float)NR_);
  int t = idx >> 7;
  int d = idx & 127;
  int i2 = (d >> 1) << 1;
  float dv = __expf((float)i2 * (-9.210340371976184f / (float)D_DIM)); // -ln(1e4)/D
  float ang = (float)t * dv;
  s += (d & 1) ? cosf(ang) : sinf(ang);
  x[idx] = s;
}

// ----------------------------- tiny MHA (L=96, H=8, dh=16) ------------------
__global__ void mha_kernel(const float* __restrict__ qkv, float* __restrict__ o) {
  int h = blockIdx.x;
  int q = threadIdx.x;
  if (q >= TT_) return;
  const float* Q = qkv + q * 384 + h * 16;
  float qv[16];
#pragma unroll
  for (int i = 0; i < 16; ++i) qv[i] = Q[i];
  float mx = -1e30f;
  for (int k = 0; k < TT_; ++k) {
    const float* Kp = qkv + k * 384 + 128 + h * 16;
    float s = 0.f;
#pragma unroll
    for (int i = 0; i < 16; ++i) s += qv[i] * Kp[i];
    s *= 0.25f;                       // 1/sqrt(16)
    mx = fmaxf(mx, s);
  }
  float den = 0.f;
  float oacc[16];
#pragma unroll
  for (int i = 0; i < 16; ++i) oacc[i] = 0.f;
  for (int k = 0; k < TT_; ++k) {
    const float* Kp = qkv + k * 384 + 128 + h * 16;
    float s = 0.f;
#pragma unroll
    for (int i = 0; i < 16; ++i) s += qv[i] * Kp[i];
    s *= 0.25f;
    float p = __expf(s - mx);
    den += p;
    const float* Vp = qkv + k * 384 + 256 + h * 16;
#pragma unroll
    for (int i = 0; i < 16; ++i) oacc[i] += p * Vp[i];
  }
  float inv = 1.f / den;
#pragma unroll
  for (int i = 0; i < 16; ++i) o[q * 128 + h * 16 + i] = oacc[i] * inv;
}

// ----------------------------- host-side helpers ----------------------------

static inline void run_fill(hipStream_t s, float* p, float v, int n) {
  fill_f32_kernel<<<(n + 255) / 256, 256, 0, s>>>(p, v, n);
}
static inline void run_fill_u(hipStream_t s, unsigned* p, unsigned v, int n) {
  fill_u32_kernel<<<(n + 255) / 256, 256, 0, s>>>(p, v, n);
}
static inline void run_gemm(hipStream_t s, const float* X, const float* W, const float* b,
                            float* Y, int M, int N, int K, int gelu) {
  dim3 g((unsigned)(N / GEMM_BN), (unsigned)((M + GEMM_BM - 1) / GEMM_BM));
  gemm_wmma_kernel<<<g, 256, 0, s>>>(X, W, b, Y, M, N, K, gelu);
}
static inline void run_ln(hipStream_t s, const float* in, const float* in_bias,
                          const float* pre, const float* g, const float* b,
                          const float* post, float* out, int rows, int gelu) {
  ln_kernel<<<(rows + 3) / 4, 128, 0, s>>>(in, in_bias, pre, g, b, post, out, rows, gelu);
}

struct GatP { const float *Wl, *Wr, *att, *bias, *bl, *br; };

// One-head-at-a-time GATv2 (bias NOT added here; fold into the following LN).
static void run_gat(hipStream_t s, const float* f, int n, const int* ei, int Eorig,
                    const GatP& p, float* out, float* xlh, float* xrh,
                    float* logit, float* ex, float* nmax, float* nden) {
  int Etot = Eorig + n;
  run_fill(s, out, 0.f, n * D_DIM);
  for (int h = 0; h < H_HEADS; ++h) {
    run_gemm(s, f, p.Wl + (size_t)h * C_DIM * D_DIM, p.bl + h * C_DIM, xlh, n, C_DIM, D_DIM, 0);
    run_gemm(s, f, p.Wr + (size_t)h * C_DIM * D_DIM, p.br + h * C_DIM, xrh, n, C_DIM, D_DIM, 0);
    run_fill_u(s, (unsigned*)nmax, 0u, n);
    run_fill(s, nden, 0.f, n);
    edge_logit_kernel<<<(Etot + 3) / 4, 128, 0, s>>>(xlh, xrh, p.att + h * C_DIM, ei,
                                                     Eorig, Etot, logit, (unsigned*)nmax);
    edge_exp_kernel<<<(Etot + 255) / 256, 256, 0, s>>>(logit, (const unsigned*)nmax, ei,
                                                       Eorig, Etot, ex, nden);
    edge_accum_kernel<<<(Etot + 3) / 4, 128, 0, s>>>(xlh, ex, nden, ei, Eorig, Etot, out);
  }
}

// ----------------------------- kernel_launch --------------------------------
//
// Input mapping assumption: d_in follows setup_inputs() dict order with the
// nested `params` dict tree-flattened in sorted-key order (jax convention):
//   [0] x  [1] edge_index  [2] node_type  [3] asset_to_zone  [4] zone_to_region
//   [5] zone_edges  [6] region_edges  [7] N_base  [8] T
//   [9..132] params leaves:
//     asset_convs[0..1]{Wl,Wr,att,bias,bl,br}, asset_dec{...}, asset_norms[0..1]{b,g},
//     fn_asset{b,g}, fn_nation{b,g}, fn_region{b,g}, fn_zone{b,g}, in_b, in_w,
//     region_convs[0..1]{...}, region_dec{...}, region_norms[0..1]{b,g},
//     tlayers[0..3]{b1,b2,in_b,in_w,ln1{b,g},ln2{b,g},out_b,out_w,w1,w2},
//     zone_convs[0..1]{...}, zone_dec{...}, zone_norms[0..1]{b,g}

extern "C" void kernel_launch(void* const* d_in, const int* in_sizes, int n_in,
                              void* d_out, int out_size, void* d_ws, size_t ws_size,
                              hipStream_t stream) {
  if (n_in < 133) return;

  const float* x_in   = (const float*)d_in[0];
  const int*   ei_a   = (const int*)d_in[1];
  const int*   a2z    = (const int*)d_in[3];
  const int*   z2r    = (const int*)d_in[4];
  const int*   ei_z   = (const int*)d_in[5];
  const int*   ei_r   = (const int*)d_in[6];
  const int Ea = in_sizes[1] / 2;   // 192000
  const int Ez = in_sizes[5] / 2;   // 20000
  const int Er = in_sizes[6] / 2;   // 2000

  const int B = 9;
  auto F = [&](int i) { return (const float*)d_in[i]; };
  auto gat_at = [&](int i) {
    GatP p = { F(i), F(i + 1), F(i + 2), F(i + 3), F(i + 4), F(i + 5) };
    return p;
  };
  GatP asset_conv[2] = { gat_at(B + 0), gat_at(B + 6) };
  GatP asset_dec     = gat_at(B + 12);
  const float* asset_norm_b[2] = { F(B + 18), F(B + 20) };
  const float* asset_norm_g[2] = { F(B + 19), F(B + 21) };
  const float *fn_asset_b = F(B + 22), *fn_asset_g = F(B + 23);
  const float *fn_nat_b   = F(B + 24), *fn_nat_g   = F(B + 25);
  const float *fn_reg_b   = F(B + 26), *fn_reg_g   = F(B + 27);
  const float *fn_zone_b  = F(B + 28), *fn_zone_g  = F(B + 29);
  const float *in_b = F(B + 30), *in_w = F(B + 31);
  GatP region_conv[2] = { gat_at(B + 32), gat_at(B + 38) };
  GatP region_dec     = gat_at(B + 44);
  const float* region_norm_b[2] = { F(B + 50), F(B + 52) };
  const float* region_norm_g[2] = { F(B + 51), F(B + 53) };
  struct TlP { const float *b1, *b2, *in_b, *in_w, *ln1b, *ln1g, *ln2b, *ln2g, *out_b, *out_w, *w1, *w2; };
  TlP tl[4];
  for (int i = 0; i < 4; ++i) {
    int TB = B + 54 + 12 * i;
    tl[i] = { F(TB + 0), F(TB + 1), F(TB + 2), F(TB + 3), F(TB + 4), F(TB + 5),
              F(TB + 6), F(TB + 7), F(TB + 8), F(TB + 9), F(TB + 10), F(TB + 11) };
  }
  GatP zone_conv[2] = { gat_at(B + 102), gat_at(B + 108) };
  GatP zone_dec     = gat_at(B + 114);
  const float* zone_norm_b[2] = { F(B + 120), F(B + 122) };
  const float* zone_norm_g[2] = { F(B + 121), F(B + 123) };

  // ----- workspace carve (fp32), ~133.8 MB total -----
  float* wsf = (float*)d_ws;
  size_t off = 0;
  auto alloc = [&](size_t n) { float* p = wsf + off; off += n; return p; };
  float* hA   = alloc((size_t)N_ASSET * D_DIM);  // asset features / skip
  float* hB   = alloc((size_t)N_ASSET * D_DIM);  // decoder input
  float* hC   = alloc((size_t)N_ASSET * D_DIM);  // GAT accumulator
  float* xlh  = alloc((size_t)N_ASSET * D_DIM);  // per-head xl
  float* xrh  = alloc((size_t)N_ASSET * D_DIM);  // per-head xr
  float* logit = alloc(Ea + N_ASSET);
  float* exb   = alloc(Ea + N_ASSET);
  float* nmax  = alloc(N_ASSET);
  float* nden  = alloc(N_ASSET);
  float* zA = alloc((size_t)N_ZONE * D_DIM);
  float* zB = alloc((size_t)N_ZONE * D_DIM);
  float* zC = alloc((size_t)N_ZONE * D_DIM);
  float* rA = alloc((size_t)N_REG * D_DIM);
  float* rB = alloc((size_t)N_REG * D_DIM);
  float* rC = alloc((size_t)N_REG * D_DIM);
  float* natX   = alloc(TD);
  float* natQKV = alloc(TT_ * 384);
  float* natO   = alloc(TD);
  float* natP   = alloc(TD);
  float* natFF  = alloc(TT_ * 512);
  float* cnt    = alloc(64);
  (void)ws_size; (void)out_size;

  float* out_assets  = (float*)d_out;
  float* out_zones   = out_assets + (size_t)NB_ * TD;
  float* out_regions = out_zones + (size_t)NZ_ * TD;
  float* out_nation  = out_regions + (size_t)NR_ * TD;

  // ===== 1. input projection: h = x @ in_w.T + in_b  [48000,32]->[48000,128]
  run_gemm(stream, x_in, in_w, in_b, hA, N_ASSET, D_DIM, FD_, 0);

  // ===== 2. asset GAT blocks (x2), in-place on hA
  for (int i = 0; i < 2; ++i) {
    run_gat(stream, hA, N_ASSET, ei_a, Ea, asset_conv[i], hC, xlh, xrh, logit, exb, nmax, nden);
    run_ln(stream, hC, asset_conv[i].bias, nullptr, asset_norm_g[i], asset_norm_b[i],
           hA, hA, N_ASSET, 1);
  }

  // ===== 3. pool assets -> zones
  run_fill(stream, cnt, 0.f, 64);
  count_kernel<<<(NB_ + 255) / 256, 256, 0, stream>>>(a2z, NB_, cnt);
  run_fill(stream, zA, 0.f, N_ZONE * D_DIM);
  pool_accum_kernel<<<((NB_ * TD) + 255) / 256, 256, 0, stream>>>(hA, a2z, zA, NB_ * TD);
  pool_div_kernel<<<((NZ_ * TD) + 255) / 256, 256, 0, stream>>>(zA, cnt, NZ_ * TD);

  // ===== 4. zone GAT blocks (x2)
  for (int i = 0; i < 2; ++i) {
    run_gat(stream, zA, N_ZONE, ei_z, Ez, zone_conv[i], zB, xlh, xrh, logit, exb, nmax, nden);
    run_ln(stream, zB, zone_conv[i].bias, nullptr, zone_norm_g[i], zone_norm_b[i],
           zA, zA, N_ZONE, 1);
  }

  // ===== 5. pool zones -> regions
  run_fill(stream, cnt, 0.f, 64);
  count_kernel<<<1, 64, 0, stream>>>(z2r, NZ_, cnt);
  run_fill(stream, rA, 0.f, N_REG * D_DIM);
  pool_accum_kernel<<<((NZ_ * TD) + 255) / 256, 256, 0, stream>>>(zA, z2r, rA, NZ_ * TD);
  pool_div_kernel<<<((NR_ * TD) + 255) / 256, 256, 0, stream>>>(rA, cnt, NR_ * TD);

  // ===== 6. region GAT blocks (x2)
  for (int i = 0; i < 2; ++i) {
    run_gat(stream, rA, N_REG, ei_r, Er, region_conv[i], rB, xlh, xrh, logit, exb, nmax, nden);
    run_ln(stream, rB, region_conv[i].bias, nullptr, region_norm_g[i], region_norm_b[i],
           rA, rA, N_REG, 1);
  }

  // ===== 7. nation: mean over regions + posenc, 4 transformer layers
  nation_init_kernel<<<(TD + 255) / 256, 256, 0, stream>>>(rA, natX);
  for (int l = 0; l < 4; ++l) {
    run_gemm(stream, natX, tl[l].in_w, tl[l].in_b, natQKV, TT_, 384, D_DIM, 0);
    mha_kernel<<<H_HEADS, TT_, 0, stream>>>(natQKV, natO);
    run_gemm(stream, natO, tl[l].out_w, tl[l].out_b, natP, TT_, D_DIM, D_DIM, 0);
    run_ln(stream, natP, nullptr, natX, tl[l].ln1g, tl[l].ln1b, nullptr, natX, TT_, 0);
    run_gemm(stream, natX, tl[l].w1, tl[l].b1, natFF, TT_, 512, D_DIM, 1);   // gelu
    run_gemm(stream, natFF, tl[l].w2, tl[l].b2, natP, TT_, D_DIM, 512, 0);
    run_ln(stream, natP, nullptr, natX, tl[l].ln2g, tl[l].ln2b, nullptr, natX, TT_, 0);
  }
  run_ln(stream, natX, nullptr, nullptr, fn_nat_g, fn_nat_b, nullptr, out_nation, TT_, 0);

  // ===== 8. region decoder
  gather_add_kernel<<<((NR_ * TD) + 255) / 256, 256, 0, stream>>>(out_nation, rA, nullptr,
                                                                  rC, NR_ * TD);
  run_gat(stream, rC, N_REG, ei_r, Er, region_dec, rB, xlh, xrh, logit, exb, nmax, nden);
  run_ln(stream, rB, region_dec.bias, nullptr, fn_reg_g, fn_reg_b, nullptr,
         out_regions, N_REG, 0);

  // ===== 9. zone decoder
  gather_add_kernel<<<((NZ_ * TD) + 255) / 256, 256, 0, stream>>>(out_regions, zA, z2r,
                                                                  zC, NZ_ * TD);
  run_gat(stream, zC, N_ZONE, ei_z, Ez, zone_dec, zB, xlh, xrh, logit, exb, nmax, nden);
  run_ln(stream, zB, zone_dec.bias, nullptr, fn_zone_g, fn_zone_b, nullptr,
         out_zones, N_ZONE, 0);

  // ===== 10. asset decoder
  gather_add_kernel<<<((NB_ * TD) + 255) / 256, 256, 0, stream>>>(out_zones, hA, a2z,
                                                                  hB, NB_ * TD);
  run_gat(stream, hB, N_ASSET, ei_a, Ea, asset_dec, hC, xlh, xrh, logit, exb, nmax, nden);
  run_ln(stream, hC, asset_dec.bias, nullptr, fn_asset_g, fn_asset_b, nullptr,
         out_assets, N_ASSET, 0);
}
